// GumbelVectorQuantizer_83124797047428
// MI455X (gfx1250) — compile-verified, matrix operand order checked by reading, not run on hardware
//
#include <hip/hip_runtime.h>

typedef __attribute__((ext_vector_type(16))) __bf16 v16bf;
typedef __attribute__((ext_vector_type(8)))  __bf16 v8bf;
typedef __attribute__((ext_vector_type(8)))  float  v8f;

#define DIM      256
#define MCODES   1024
#define NROWS    16
#define NTOT     32768                 // BSZ*TSZ
#define NBLOCKS  (NTOT / NROWS)        // 2048

// output layout (reference tuple, flattened in return order)
#define CODE_OFF 8388608               // NTOT*DIM
#define PROB_OFF 8388609
#define INDS_OFF 8388610
#define LOSS_OFF 8421378               // INDS_OFF + NTOT

// workspace layout (bytes)
#define WS_EMBB   0                    // MCODES*DIM*2 = 524288
#define WS_ENORM  524288               // MCODES*4
#define WS_AVGP   528384               // MCODES*4
#define WS_CNT    532480               // MCODES*4
#define WS_LOSS   536576               // 4

// ---------------------------------------------------------------------------
// Prep: emb (f32) -> bf16 copy + ||e||^2 ; zero all accumulators every call.
// ---------------------------------------------------------------------------
__global__ __launch_bounds__(256) void vq_prep(const float* __restrict__ emb,
                                               __bf16* __restrict__ embb,
                                               float* __restrict__ enorm2,
                                               float* __restrict__ avg_probs,
                                               float* __restrict__ counts,
                                               float* __restrict__ loss_accum) {
    __shared__ float sc[256];
    const int m   = blockIdx.x;        // code row 0..1023
    const int tid = threadIdx.x;       // dim 0..255
    float v = emb[(size_t)m * DIM + tid];
    embb[(size_t)m * DIM + tid] = (__bf16)v;
    sc[tid] = v * v;
    __syncthreads();
    for (int off = 128; off > 0; off >>= 1) {
        if (tid < off) sc[tid] += sc[tid + off];
        __syncthreads();
    }
    if (tid == 0) {
        enorm2[m]    = sc[0];
        avg_probs[m] = 0.0f;
        counts[m]    = 0.0f;
        if (m == 0) loss_accum[0] = 0.0f;
    }
}

// ---------------------------------------------------------------------------
// Main: per block, 16 rows of x vs all 1024 codes. 8 waves; each wave owns
// 8 column tiles of 16 codes; logits live entirely in registers.
// ---------------------------------------------------------------------------
__global__ __launch_bounds__(256) void vq_main(const float*  __restrict__ x,
                                               const float*  __restrict__ emb,
                                               const __bf16* __restrict__ embb,
                                               const float*  __restrict__ enorm2,
                                               float* __restrict__ quantized,
                                               float* __restrict__ inds_out,
                                               float* __restrict__ avg_probs,
                                               float* __restrict__ counts,
                                               float* __restrict__ loss_accum) {
    __shared__ __align__(16) __bf16 xb[NROWS][DIM];
    __shared__ float xn2[NROWS];
    __shared__ float sc[NROWS][16];     // generic 16x16 reduction scratch
    __shared__ float wred[8][NROWS];    // per-wave row partials
    __shared__ int   wredi[8][NROWS];
    __shared__ float rmax[NROWS], rsinv[NROWS];
    __shared__ int   rarg[NROWS];

    const int tid  = threadIdx.x;
    const int row0 = blockIdx.x * NROWS;
    const int lane = tid & 31;
    const int wv   = tid >> 5;          // wave 0..7
    const int half = lane >> 4;         // 0/1
    const int l16  = lane & 15;

    // ---- load x tile -> bf16 LDS, row norms ------------------------------
    {
        const int r = tid >> 4, sub = tid & 15;
        const float* xr = x + (size_t)(row0 + r) * DIM;
        float s = 0.0f;
        #pragma unroll
        for (int j = 0; j < 16; ++j) {
            int c = j * 16 + sub;                 // coalesced
            float v = xr[c];
            xb[r][c] = (__bf16)v;
            s += v * v;
        }
        sc[r][sub] = s;
    }
    __syncthreads();
    if (tid < NROWS) {
        float s = 0.0f;
        #pragma unroll
        for (int j = 0; j < 16; ++j) s += sc[tid][j];
        xn2[tid] = s;
    }
    __syncthreads();

    // ---- A fragments (x rows), reused across all column tiles ------------
    union ABu { v16bf v; v8bf h[2]; };
    v16bf afrag[8];
    #pragma unroll
    for (int kc = 0; kc < 8; ++kc) {
        const __bf16* src = &xb[l16][kc * 32 + half * 8];
        ABu u;
        u.h[0] = *(const v8bf*)(src);
        u.h[1] = *(const v8bf*)(src + 16);
        afrag[kc] = u.v;
    }
    float xnr[8];
    #pragma unroll
    for (int v = 0; v < 8; ++v) xnr[v] = xn2[half * 8 + v];

    // ---- WMMA: 8 tiles x (K=256 in 8 chunks); logits -> registers --------
    v8f cacc[8];
    #pragma unroll
    for (int t = 0; t < 8; ++t) {
        const int cb = (wv * 8 + t) * 16;         // code tile base
        const __bf16* eb = embb + (size_t)(cb + l16) * DIM + half * 8;
        v8f acc = {};
        #pragma unroll
        for (int kc = 0; kc < 8; ++kc) {
            ABu u;
            u.h[0] = *(const v8bf*)(eb + kc * 32);
            u.h[1] = *(const v8bf*)(eb + kc * 32 + 16);
            acc = __builtin_amdgcn_wmma_f32_16x16x32_bf16(
                false, afrag[kc], false, u.v, (short)0, acc, false, false);
        }
        const float en = enorm2[cb + l16];
        #pragma unroll
        for (int v = 0; v < 8; ++v)               // logits = -distance
            acc[v] = 2.0f * acc[v] - en - xnr[v];
        cacc[t] = acc;
    }

    // ---- per-row max + argmax (first-index tie-break) --------------------
    float mx[8]; int ax[8];
    #pragma unroll
    for (int v = 0; v < 8; ++v) { mx[v] = -3.402823466e38f; ax[v] = 0; }
    #pragma unroll
    for (int t = 0; t < 8; ++t) {
        const int col = (wv * 8 + t) * 16 + l16;
        #pragma unroll
        for (int v = 0; v < 8; ++v) {
            float val = cacc[t][v];
            if (val > mx[v]) { mx[v] = val; ax[v] = col; }
        }
    }
    #pragma unroll
    for (int off = 8; off >= 1; off >>= 1) {
        #pragma unroll
        for (int v = 0; v < 8; ++v) {
            float ov = __shfl_xor(mx[v], off, 32);
            int   oi = __shfl_xor(ax[v], off, 32);
            if (ov > mx[v] || (ov == mx[v] && oi < ax[v])) { mx[v] = ov; ax[v] = oi; }
        }
    }
    if (l16 == 0) {
        #pragma unroll
        for (int v = 0; v < 8; ++v) {
            wred [wv][half * 8 + v] = mx[v];
            wredi[wv][half * 8 + v] = ax[v];
        }
    }
    __syncthreads();
    if (tid < NROWS) {
        float m = wred[0][tid]; int a = wredi[0][tid];
        #pragma unroll
        for (int w = 1; w < 8; ++w) {
            float ov = wred[w][tid]; int oi = wredi[w][tid];
            if (ov > m || (ov == m && oi < a)) { m = ov; a = oi; }
        }
        rmax[tid] = m; rarg[tid] = a;
    }
    __syncthreads();

    // ---- per-row sum(exp) ------------------------------------------------
    float rmr[8];
    #pragma unroll
    for (int v = 0; v < 8; ++v) rmr[v] = rmax[half * 8 + v];
    float sm[8];
    #pragma unroll
    for (int v = 0; v < 8; ++v) sm[v] = 0.0f;
    #pragma unroll
    for (int t = 0; t < 8; ++t)
        #pragma unroll
        for (int v = 0; v < 8; ++v) sm[v] += __expf(cacc[t][v] - rmr[v]);
    #pragma unroll
    for (int off = 8; off >= 1; off >>= 1)
        #pragma unroll
        for (int v = 0; v < 8; ++v) sm[v] += __shfl_xor(sm[v], off, 32);
    if (l16 == 0) {
        #pragma unroll
        for (int v = 0; v < 8; ++v) wred[wv][half * 8 + v] = sm[v];
    }
    __syncthreads();
    if (tid < NROWS) {
        float s = 0.0f;
        #pragma unroll
        for (int w = 0; w < 8; ++w) s += wred[w][tid];
        rsinv[tid] = 1.0f / s;
    }
    __syncthreads();

    // ---- avg_probs: column sums over the 16 rows, one atomic per column --
    float rir[8];
    #pragma unroll
    for (int v = 0; v < 8; ++v) rir[v] = rsinv[half * 8 + v];
    #pragma unroll
    for (int t = 0; t < 8; ++t) {
        const int col = (wv * 8 + t) * 16 + l16;
        float p = 0.0f;
        #pragma unroll
        for (int v = 0; v < 8; ++v) p += __expf(cacc[t][v] - rmr[v]) * rir[v];
        p += __shfl_xor(p, 16, 32);               // combine both row-halves
        if (half == 0) atomicAdd(&avg_probs[col], p);
    }

    // ---- gather quantized, commitment loss, inds, histogram --------------
    {
        const int r = tid >> 4, sub = tid & 15;
        const int k = rarg[r];
        const float* er = emb + (size_t)k * DIM;
        const float* xr = x + (size_t)(row0 + r) * DIM;
        float* qr = quantized + (size_t)(row0 + r) * DIM;
        float ls = 0.0f;
        #pragma unroll
        for (int j = 0; j < 16; ++j) {
            int c = j * 16 + sub;                 // coalesced
            float e = er[c];
            qr[c] = e;
            float d = xr[c] - e;
            ls += d * d;
        }
        sc[r][sub] = ls;
    }
    __syncthreads();
    if (tid < NROWS) {
        float s = 0.0f;
        #pragma unroll
        for (int j = 0; j < 16; ++j) s += sc[tid][j];
        sc[tid][0] = s;
        inds_out[row0 + tid] = (float)rarg[tid];
        atomicAdd(&counts[rarg[tid]], 1.0f);
    }
    __syncthreads();
    if (tid == 0) {
        float s = 0.0f;
        #pragma unroll
        for (int r = 0; r < NROWS; ++r) s += sc[r][0];
        atomicAdd(loss_accum, s);
    }
}

// ---------------------------------------------------------------------------
// Finalize: entropies + loss normalization -> 3 output scalars.
// ---------------------------------------------------------------------------
__global__ __launch_bounds__(256) void vq_finalize(const float* __restrict__ avg_probs,
                                                   const float* __restrict__ counts,
                                                   const float* __restrict__ loss_accum,
                                                   float* __restrict__ out) {
    __shared__ float s1[256], s2[256];
    const int tid = threadIdx.x;
    const float invN = 1.0f / (float)NTOT;
    float ce = 0.0f, pe = 0.0f;
    #pragma unroll
    for (int j = 0; j < 4; ++j) {
        int m = tid * 4 + j;
        float hp = counts[m] * invN;
        ce -= hp * log2f(hp + 1e-10f);
        float ap = avg_probs[m] * invN;
        pe -= ap * log2f(ap + 1e-10f);
    }
    s1[tid] = ce; s2[tid] = pe;
    __syncthreads();
    for (int off = 128; off > 0; off >>= 1) {
        if (tid < off) { s1[tid] += s1[tid + off]; s2[tid] += s2[tid + off]; }
        __syncthreads();
    }
    if (tid == 0) {
        out[CODE_OFF] = s1[0];
        out[PROB_OFF] = s2[0];
        out[LOSS_OFF] = loss_accum[0] / (float)((size_t)NTOT * DIM);
    }
}

// ---------------------------------------------------------------------------
extern "C" void kernel_launch(void* const* d_in, const int* in_sizes, int n_in,
                              void* d_out, int out_size, void* d_ws, size_t ws_size,
                              hipStream_t stream) {
    const float* x   = (const float*)d_in[0];   // [16,2048,256]
    const float* emb = (const float*)d_in[1];   // [1,1024,256]

    char* ws = (char*)d_ws;
    __bf16* embb      = (__bf16*)(ws + WS_EMBB);
    float*  enorm2    = (float*)(ws + WS_ENORM);
    float*  avg_probs = (float*)(ws + WS_AVGP);
    float*  counts    = (float*)(ws + WS_CNT);
    float*  loss      = (float*)(ws + WS_LOSS);

    float* out       = (float*)d_out;
    float* quantized = out;                     // [NTOT, DIM]
    float* inds      = out + INDS_OFF;          // [NTOT] as float

    vq_prep<<<MCODES, 256, 0, stream>>>(emb, embb, enorm2, avg_probs, counts, loss);
    vq_main<<<NBLOCKS, 256, 0, stream>>>(x, emb, embb, enorm2,
                                         quantized, inds, avg_probs, counts, loss);
    vq_finalize<<<1, 256, 0, stream>>>(avg_probs, counts, loss, out);
}
